// SelfAttention_33698313404907
// MI455X (gfx1250) — matrix operand order repaired and checked by reading, not compiled, not run on hardware
//
#include <hip/hip_runtime.h>
#include <hip/hip_bf16.h>

// ---------------------------------------------------------------------------
// Fused self-attention for MI455X (gfx1250, wave32, WMMA, async-to-LDS).
//   K0: one-shot f32 -> bf16 conversion of x and the four weight matrices
//   K1: fused QKV projection (bf16 WMMA, one A-frag feeds 6 WMMAs)
//   K2: flash attention with double-buffered async LDS staging of K/V tiles
//   K3: output projection + LayerNorm + LeakyReLU
// A/B fragments use contiguous k order (k = 16*half + s, same in A and B); f32 C/D as in the ISA.
// ---------------------------------------------------------------------------

typedef __attribute__((ext_vector_type(16))) __bf16 v16bf;
typedef __attribute__((ext_vector_type(8)))  __bf16 v8bf;
typedef __attribute__((ext_vector_type(8)))  float  v8f;

#define BATCH  8
#define SEQ    2048
#define DIM    256
#define LN_EPS 1e-5f
#define SLOPE  0.01f

static __device__ __forceinline__ v8f wmma_bf16(v16bf a, v16bf b, v8f c) {
  return __builtin_amdgcn_wmma_f32_16x16x32_bf16(false, a, false, b, (short)0, c,
                                                 false, false);
}

// Two contiguous 8-bf16 chunks -> v16bf A fragment (lane = row, K-chunks kh*16 / kh*16+8)
static __device__ __forceinline__ v16bf load_a_bf16(const __bf16* lo, const __bf16* hi) {
  v8bf a = *(const v8bf*)lo;
  v8bf b = *(const v8bf*)hi;
  v16bf r;
#pragma unroll
  for (int i = 0; i < 8; ++i) { r[i] = a[i]; r[i + 8] = b[i]; }
  return r;
}

// Async global->LDS copy of 16 bytes (ASYNCcnt-tracked, no VGPR data path).
static __device__ __forceinline__ void async_b128(unsigned lds_off, const void* gptr) {
  asm volatile("global_load_async_to_lds_b128 %0, %1, off"
               :: "v"(lds_off), "v"((unsigned long long)(uintptr_t)gptr)
               : "memory");
}

// ---------------------------------------------------------------------------
// Kernel 0: f32 -> bf16 elementwise conversion (8 elems / thread).
// ---------------------------------------------------------------------------
__global__ __launch_bounds__(256)
void cvt_x_kernel(const float* __restrict__ src, __bf16* __restrict__ dst) {
  const size_t i = ((size_t)blockIdx.x * 256 + threadIdx.x) * 8;
  float4 a = *(const float4*)(src + i);
  float4 b = *(const float4*)(src + i + 4);
  v8bf r;
  r[0] = (__bf16)a.x; r[1] = (__bf16)a.y; r[2] = (__bf16)a.z; r[3] = (__bf16)a.w;
  r[4] = (__bf16)b.x; r[5] = (__bf16)b.y; r[6] = (__bf16)b.z; r[7] = (__bf16)b.w;
  *(v8bf*)(dst + i) = r;
}

__global__ __launch_bounds__(256)
void cvt_w_kernel(const float* __restrict__ w0, const float* __restrict__ w1,
                  const float* __restrict__ w2, const float* __restrict__ w3,
                  __bf16* __restrict__ o0, __bf16* __restrict__ o1,
                  __bf16* __restrict__ o2, __bf16* __restrict__ o3) {
  const float* src;
  __bf16* dst;
  switch (blockIdx.y) {
    case 0:  src = w0; dst = o0; break;
    case 1:  src = w1; dst = o1; break;
    case 2:  src = w2; dst = o2; break;
    default: src = w3; dst = o3; break;
  }
  const size_t i = ((size_t)blockIdx.x * 256 + threadIdx.x) * 8;
  float4 a = *(const float4*)(src + i);
  float4 b = *(const float4*)(src + i + 4);
  v8bf r;
  r[0] = (__bf16)a.x; r[1] = (__bf16)a.y; r[2] = (__bf16)a.z; r[3] = (__bf16)a.w;
  r[4] = (__bf16)b.x; r[5] = (__bf16)b.y; r[6] = (__bf16)b.z; r[7] = (__bf16)b.w;
  *(v8bf*)(dst + i) = r;
}

// ---------------------------------------------------------------------------
// Kernel 1: fused Q/K/V projection.  Block = 8 waves, 16 rows x 256 cols;
// wave w owns cols [32w,32w+32).  One A-fragment feeds 6 WMMAs (Q0,Q1,K0,K1,V0,V1).
// ---------------------------------------------------------------------------
__global__ __launch_bounds__(256)
void qkv_proj_kernel(const __bf16* __restrict__ Xb,
                     const __bf16* __restrict__ Wqb, const float* __restrict__ bq,
                     const __bf16* __restrict__ Wkb, const float* __restrict__ bk,
                     const __bf16* __restrict__ Wvb, const float* __restrict__ bv,
                     __bf16* __restrict__ Qb, __bf16* __restrict__ Kb,
                     __bf16* __restrict__ Vt) {
  const int lane = threadIdx.x & 31;
  const int wave = threadIdx.x >> 5;
  const int half = lane >> 4;
  const int l16  = lane & 15;
  const int row0 = blockIdx.x * 16;
  const int col0 = wave * 32;

  const __bf16* xrow = Xb + (size_t)(row0 + l16) * DIM;
  const size_t wro0 = (size_t)(col0 + l16) * DIM;       // B-frag row (tile 0)
  const size_t wro1 = (size_t)(col0 + 16 + l16) * DIM;  // B-frag row (tile 1)

  v8f cq0 = {}, cq1 = {}, ck0 = {}, ck1 = {}, cv0 = {}, cv1 = {};
#pragma unroll
  for (int k0 = 0; k0 < DIM; k0 += 32) {
    v16bf a = load_a_bf16(xrow + k0 + half * 16, xrow + k0 + half * 16 + 8);
    const int kb = k0 + half * 16;
    v16bf q0 = *(const v16bf*)(Wqb + wro0 + kb);
    v16bf q1 = *(const v16bf*)(Wqb + wro1 + kb);
    v16bf k0v = *(const v16bf*)(Wkb + wro0 + kb);
    v16bf k1v = *(const v16bf*)(Wkb + wro1 + kb);
    v16bf v0 = *(const v16bf*)(Wvb + wro0 + kb);
    v16bf v1 = *(const v16bf*)(Wvb + wro1 + kb);
    cq0 = wmma_bf16(a, q0, cq0);
    cq1 = wmma_bf16(a, q1, cq1);
    ck0 = wmma_bf16(a, k0v, ck0);
    ck1 = wmma_bf16(a, k1v, ck1);
    cv0 = wmma_bf16(a, v0, cv0);
    cv1 = wmma_bf16(a, v1, cv1);
  }

  const float bq0 = bq[col0 + l16], bq1 = bq[col0 + 16 + l16];
  const float bk0 = bk[col0 + l16], bk1 = bk[col0 + 16 + l16];
  const float bv0 = bv[col0 + l16], bv1 = bv[col0 + 16 + l16];
#pragma unroll
  for (int v = 0; v < 8; ++v) {  // C layout: VGPR v -> row v + 8*half, col = l16
    const int r = row0 + v + half * 8;
    Qb[(size_t)r * DIM + col0 + l16]      = (__bf16)(cq0[v] + bq0);
    Qb[(size_t)r * DIM + col0 + 16 + l16] = (__bf16)(cq1[v] + bq1);
    Kb[(size_t)r * DIM + col0 + l16]      = (__bf16)(ck0[v] + bk0);
    Kb[(size_t)r * DIM + col0 + 16 + l16] = (__bf16)(ck1[v] + bk1);
    const int bb = r >> 11;  // batch
    const int n  = r & 2047; // seq pos
    Vt[(size_t)bb * DIM * SEQ + (size_t)(col0 + l16) * SEQ + n]      = (__bf16)(cv0[v] + bv0);
    Vt[(size_t)bb * DIM * SEQ + (size_t)(col0 + 16 + l16) * SEQ + n] = (__bf16)(cv1[v] + bv1);
  }
}

// ---------------------------------------------------------------------------
// Kernel 2: flash attention with async double-buffered K/V staging in LDS.
// Block = 8 waves; wave owns 16 query rows (block: 128 rows of one batch).
// Per 32-key step: stage next K/V tile async while computing current tile.
// ---------------------------------------------------------------------------
__global__ __launch_bounds__(256)
void flash_attn_kernel(const __bf16* __restrict__ Qb, const __bf16* __restrict__ Kb,
                       const __bf16* __restrict__ Vt, __bf16* __restrict__ Loc) {
  __shared__ __bf16 kts[2][32][DIM];   // K tile: [key][d]      (2 x 16 KB)
  __shared__ __bf16 vts[2][DIM][32];   // V tile: [d][key]      (2 x 16 KB)
  __shared__ __bf16 pbuf[8][16][32];   // per-wave P transpose  (8 KB)

  const int tid  = threadIdx.x;
  const int lane = tid & 31;
  const int wave = tid >> 5;
  const int half = lane >> 4;
  const int l16  = lane & 15;
  const int b    = blockIdx.y;
  const int q0   = blockIdx.x * 128 + wave * 16;

  const __bf16* Kg = Kb + (size_t)b * SEQ * DIM;
  const __bf16* Vg = Vt + (size_t)b * DIM * SEQ;

  // ---- cooperative async staging: thread copies 64B of K + 64B of V -------
  //   K: key = tid/8 (0..31), d-chunk = (tid%8)*32 ;  V: d = tid, 32 keys
  const int skey = tid >> 3;
  const int sdch = (tid & 7) * 32;
#define STAGE_KV(buf, j)                                                        \
  {                                                                             \
    const __bf16* kg = Kg + (size_t)((j) + skey) * DIM + sdch;                  \
    unsigned klds = (unsigned)(uintptr_t)&kts[buf][skey][sdch];                 \
    const __bf16* vg = Vg + (size_t)tid * SEQ + (j);                            \
    unsigned vlds = (unsigned)(uintptr_t)&vts[buf][tid][0];                     \
    async_b128(klds,      kg);                                                  \
    async_b128(klds + 16, kg + 8);                                              \
    async_b128(klds + 32, kg + 16);                                             \
    async_b128(klds + 48, kg + 24);                                             \
    async_b128(vlds,      vg);                                                  \
    async_b128(vlds + 16, vg + 8);                                              \
    async_b128(vlds + 32, vg + 16);                                             \
    async_b128(vlds + 48, vg + 24);                                             \
  }

  // ---- preload Q A-fragments for this wave's 16 rows (8 d-chunks of 32) ----
  const __bf16* qrow = Qb + ((size_t)b * SEQ + q0 + l16) * DIM;
  v16bf qf[8];
#pragma unroll
  for (int dk = 0; dk < 8; ++dk)
    qf[dk] = load_a_bf16(qrow + dk * 32 + half * 16, qrow + dk * 32 + half * 16 + 8);

  v8f o[16];
#pragma unroll
  for (int t = 0; t < 16; ++t) o[t] = (v8f){};
  float m[8], lsum[8];
#pragma unroll
  for (int v = 0; v < 8; ++v) { m[v] = -3.0e38f; lsum[v] = 0.0f; }

  STAGE_KV(0, 0);  // prime the pipeline

  for (int j0 = 0; j0 < SEQ; j0 += 32) {
    const int cb = (j0 >> 5) & 1;
    if (j0 + 32 < SEQ) {
      STAGE_KV(cb ^ 1, j0 + 32);
      // 8 loads in flight for the next tile; wait until current tile's 8 done
      asm volatile("s_wait_asynccnt 0x8" ::: "memory");
    } else {
      asm volatile("s_wait_asynccnt 0x0" ::: "memory");
    }
    __syncthreads();  // all waves' staged data visible

    // ---- S = Q K^T for 32 keys (two 16x16 tiles), K-dim loop over D=256 ----
    v8f s0 = {}; v8f s1 = {};
#pragma unroll
    for (int dk = 0; dk < 8; ++dk) {
      v16bf k0 = *(const v16bf*)&kts[cb][l16][dk * 32 + half * 16];
      v16bf k1 = *(const v16bf*)&kts[cb][16 + l16][dk * 32 + half * 16];
      s0 = wmma_bf16(qf[dk], k0, s0);
      s1 = wmma_bf16(qf[dk], k1, s1);
    }

    // ---- online softmax (row stats replicated across each 16-lane group) ---
#pragma unroll
    for (int v = 0; v < 8; ++v) {
      float mx = fmaxf(s0[v], s1[v]);
#pragma unroll
      for (int off = 1; off < 16; off <<= 1) mx = fmaxf(mx, __shfl_xor(mx, off, 32));
      const float mn = fmaxf(m[v], mx);
      const float al = __expf(m[v] - mn);
      const float p0 = __expf(s0[v] - mn);
      const float p1 = __expf(s1[v] - mn);
      float rs = p0 + p1;
#pragma unroll
      for (int off = 1; off < 16; off <<= 1) rs += __shfl_xor(rs, off, 32);
      lsum[v] = lsum[v] * al + rs;
      m[v] = mn;
      pbuf[wave][v + half * 8][l16]      = (__bf16)p0;
      pbuf[wave][v + half * 8][16 + l16] = (__bf16)p1;
#pragma unroll
      for (int t = 0; t < 16; ++t) o[t][v] *= al;  // rescale running O
    }

    // ---- P (C layout) -> A fragment via per-wave LDS tile ------------------
    v16bf pA = load_a_bf16(&pbuf[wave][l16][half * 16],
                           &pbuf[wave][l16][half * 16 + 8]);

    // ---- O += P V, V fragments from the staged (transposed) tile -----------
#pragma unroll
    for (int t = 0; t < 16; ++t) {
      v16bf vf = *(const v16bf*)&vts[cb][t * 16 + l16][half * 16];
      o[t] = wmma_bf16(pA, vf, o[t]);
    }

    __syncthreads();  // everyone done with buffer cb before it is re-staged
  }

  // ---- normalize and write loc (bf16 row-major, feeds K3's A fragments) ----
#pragma unroll
  for (int v = 0; v < 8; ++v) {
    const float inv = 1.0f / lsum[v];
    const int r = q0 + v + half * 8;
    __bf16* out = Loc + ((size_t)b * SEQ + r) * DIM;
#pragma unroll
    for (int t = 0; t < 16; ++t) out[t * 16 + l16] = (__bf16)(o[t][v] * inv);
  }
#undef STAGE_KV
}

// ---------------------------------------------------------------------------
// Kernel 3: h = loc @ Wo^T + bo -> LayerNorm(256) -> LeakyReLU -> f32 out.
// ---------------------------------------------------------------------------
__global__ __launch_bounds__(256)
void out_proj_ln_kernel(const __bf16* __restrict__ Loc,
                        const __bf16* __restrict__ Wob, const float* __restrict__ bo,
                        const float* __restrict__ gamma, const float* __restrict__ beta,
                        float* __restrict__ out) {
  __shared__ float hbuf[16][DIM + 4];

  const int lane = threadIdx.x & 31;
  const int wave = threadIdx.x >> 5;
  const int half = lane >> 4;
  const int l16  = lane & 15;
  const int row0 = blockIdx.x * 16;
  const int col0 = wave * 32;

  const __bf16* arow = Loc + (size_t)(row0 + l16) * DIM;
  const size_t wro0 = (size_t)(col0 + l16) * DIM;
  const size_t wro1 = (size_t)(col0 + 16 + l16) * DIM;

  v8f c0 = {}, c1 = {};
#pragma unroll
  for (int k0 = 0; k0 < DIM; k0 += 32) {
    v16bf a  = load_a_bf16(arow + k0 + half * 16, arow + k0 + half * 16 + 8);
    v16bf b0 = *(const v16bf*)(Wob + wro0 + k0 + half * 16);
    v16bf b1 = *(const v16bf*)(Wob + wro1 + k0 + half * 16);
    c0 = wmma_bf16(a, b0, c0);
    c1 = wmma_bf16(a, b1, c1);
  }

  const float bia0 = bo[col0 + l16];
  const float bia1 = bo[col0 + 16 + l16];
#pragma unroll
  for (int v = 0; v < 8; ++v) {
    hbuf[v + half * 8][col0 + l16]      = c0[v] + bia0;
    hbuf[v + half * 8][col0 + 16 + l16] = c1[v] + bia1;
  }
  __syncthreads();

  // LayerNorm: wave w reduces rows 2w (lanes 0-15) and 2w+1 (lanes 16-31).
  const int row = wave * 2 + half;
  float s = 0.0f, s2 = 0.0f;
#pragma unroll
  for (int j = 0; j < 16; ++j) {
    const float h = hbuf[row][l16 + j * 16];
    s += h; s2 += h * h;
  }
#pragma unroll
  for (int off = 1; off < 16; off <<= 1) {
    s  += __shfl_xor(s,  off, 32);
    s2 += __shfl_xor(s2, off, 32);
  }
  const float mu   = s * (1.0f / DIM);
  const float var  = s2 * (1.0f / DIM) - mu * mu;
  const float rstd = rsqrtf(var + LN_EPS);

  float* orow = out + (size_t)(row0 + row) * DIM;
#pragma unroll
  for (int j = 0; j < 16; ++j) {
    const int c = l16 + j * 16;
    const float h = (hbuf[row][c] - mu) * rstd * gamma[c] + beta[c];
    orow[c] = (h >= 0.0f) ? h : SLOPE * h;
  }
}

// ---------------------------------------------------------------------------
extern "C" void kernel_launch(void* const* d_in, const int* in_sizes, int n_in,
                              void* d_out, int out_size, void* d_ws, size_t ws_size,
                              hipStream_t stream) {
  const float* x  = (const float*)d_in[0];
  const float* Wq = (const float*)d_in[1];
  const float* bq = (const float*)d_in[2];
  const float* Wk = (const float*)d_in[3];
  const float* bk = (const float*)d_in[4];
  const float* Wv = (const float*)d_in[5];
  const float* bv = (const float*)d_in[6];
  const float* Wo = (const float*)d_in[7];
  const float* bo = (const float*)d_in[8];
  const float* g  = (const float*)d_in[9];
  const float* bt = (const float*)d_in[10];
  float* out = (float*)d_out;

  char* ws = (char*)d_ws;
  __bf16* Xb  = (__bf16*)(ws);                          // 8 MiB: x    bf16
  __bf16* Qb  = (__bf16*)(ws + ((size_t)8  << 20));     // 8 MiB: Q    [b*n][d]
  __bf16* Kb  = (__bf16*)(ws + ((size_t)16 << 20));     // 8 MiB: K    [b*n][d]
  __bf16* Vt  = (__bf16*)(ws + ((size_t)24 << 20));     // 8 MiB: V^T  [b][d][n]
  __bf16* Loc = (__bf16*)(ws + ((size_t)32 << 20));     // 8 MiB: loc  [b*n][d]
  __bf16* Wqb = (__bf16*)(ws + ((size_t)40 << 20));     // 128 KiB each
  __bf16* Wkb = Wqb + DIM * DIM;
  __bf16* Wvb = Wkb + DIM * DIM;
  __bf16* Wob = Wvb + DIM * DIM;

  // K0: convert activations (4M elems) and weights (4 x 64K elems) to bf16.
  cvt_x_kernel<<<dim3((BATCH * SEQ * DIM) / (256 * 8)), 256, 0, stream>>>(x, Xb);
  cvt_w_kernel<<<dim3((DIM * DIM) / (256 * 8), 4), 256, 0, stream>>>(
      Wq, Wk, Wv, Wo, Wqb, Wkb, Wvb, Wob);

  // K1: fused QKV projection.
  qkv_proj_kernel<<<dim3((BATCH * SEQ) / 16), 256, 0, stream>>>(
      Xb, Wqb, bq, Wkb, bk, Wvb, bv, Qb, Kb, Vt);

  // K2: flash attention.
  flash_attn_kernel<<<dim3(SEQ / 128, BATCH), 256, 0, stream>>>(Qb, Kb, Vt, Loc);

  // K3: output projection + LayerNorm + LeakyReLU.
  out_proj_ln_kernel<<<dim3((BATCH * SEQ) / 16), 256, 0, stream>>>(
      Loc, Wob, bo, g, bt, out);
}
